// SpatialCrossAttention_36919538876798
// MI455X (gfx1250) — matrix operand order repaired and checked by reading, not compiled, not run on hardware
//
#include <hip/hip_runtime.h>
#include <hip/hip_bf16.h>
#include <stddef.h>

// ---------- problem constants ----------
#define SCAMS 6
#define NQ 6400
#define CDIM 128
#define DZ 4
#define NHEADS 4
#define NPOINTS 20
#define HS_ 32
#define WS_ 88
#define MTOT (HS_ * WS_)        // 2816 (multiple of 16 -> tiles never straddle cameras)
#define DH 32                   // CDIM / NHEADS
#define KPAD 136                // padded LDS stride (elems): 272B, 16B-aligned,
                                // 4-dword lane stride mod 64 banks -> conflict-free

typedef __bf16 v16bf __attribute__((ext_vector_type(16)));
typedef __bf16 v8bf  __attribute__((ext_vector_type(8)));
typedef float  v8f   __attribute__((ext_vector_type(8)));

// pack two f32 -> two bf16 in one dword
__device__ __forceinline__ unsigned pack2bf(float a, float b) {
#if __has_builtin(__builtin_amdgcn_cvt_pk_bf16_f32)
  typedef __bf16 v2bf __attribute__((ext_vector_type(2)));
  v2bf t = __builtin_amdgcn_cvt_pk_bf16_f32(a, b);
  unsigned r; __builtin_memcpy(&r, &t, 4);
  return r;
#else
  union { float f; unsigned u; } x, y;
  x.f = a; y.f = b;
  return (x.u >> 16) | (y.u & 0xFFFF0000u);   // truncate-pack
#endif
}

__device__ __forceinline__ v8f wmma_bf16(v16bf a, v16bf b, v8f c) {
  return __builtin_amdgcn_wmma_f32_16x16x32_bf16(false, a, false, b, (short)0, c,
                                                 false, false);
}

// Fragment from LDS: two 16B-aligned ds_load_b128 at p and p+elemOff.
// A-frag: K-runs kbase..+7 and kbase+16..+23  -> elemOff = 16
// B-frag: same K-runs as A-frag               -> elemOff = 16
__device__ __forceinline__ v16bf frag_lds(const __bf16* __restrict__ p, int elemOff) {
  v8bf lo = *(const v8bf*)p;
  v8bf hi = *(const v8bf*)(p + elemOff);
  return __builtin_shufflevector(lo, hi, 0, 1, 2, 3, 4, 5, 6, 7,
                                 8, 9, 10, 11, 12, 13, 14, 15);
}

// 128-deep K accumulation with two independent WMMA chains (breaks D->C RAW
// back-to-back stalls); aRow/bCol are padded-LDS base pointers.
__device__ __forceinline__ v8f gemm_k128(const __bf16* __restrict__ aRow,
                                         const __bf16* __restrict__ bCol,
                                         int khalf) {
  v8f acc0 = {}, acc1 = {};
#pragma unroll
  for (int kk = 0; kk < 4; ++kk) {
    const int k0 = kk * 32;
    v16bf a = frag_lds(aRow + k0 + khalf * 8, 16);
    v16bf b = frag_lds(bCol + k0 + khalf * 8, 16);
    if (kk & 1) acc1 = wmma_bf16(a, b, acc1);
    else        acc0 = wmma_bf16(a, b, acc0);
  }
  return acc0 + acc1;
}

// ============================================================================
// Kernel A: val = value @ Wv + bv -> (S, HEADS, M, DH) images
// ============================================================================
__global__ void __launch_bounds__(256)
sca_val_proj(const float* __restrict__ value,   // (S, M, 1, C)
             const float* __restrict__ Wv,      // (C, C) row-major
             const float* __restrict__ bv,      // (C)
             float* __restrict__ val_ws) {      // (S, HEADS, M, DH)
  __shared__ __bf16 sWT[CDIM * KPAD];           // W^T bf16, padded (~34.8 KB)
  __shared__ __bf16 sA[16 * KPAD];              // A tile bf16, padded (~4.3 KB)
  const int r0 = blockIdx.x * 16;

  for (int i = threadIdx.x; i < CDIM * (CDIM / 2); i += 256) {
    const int c  = i & (CDIM - 1);              // coalesced over c
    const int kp = i >> 7;
    const unsigned p = pack2bf(Wv[(2 * kp) * CDIM + c], Wv[(2 * kp + 1) * CDIM + c]);
    __builtin_memcpy(&sWT[c * KPAD + 2 * kp], &p, 4);
  }
  for (int i = threadIdx.x; i < 16 * (CDIM / 2); i += 256) {
    const int r  = i >> 6;                      // 0..15
    const int kp = i & 63;
    const float2 f = *(const float2*)&value[(size_t)(r0 + r) * CDIM + 2 * kp];
    const unsigned p = pack2bf(f.x, f.y);
    __builtin_memcpy(&sA[r * KPAD + 2 * kp], &p, 4);
  }
  __syncthreads();

  const int wave  = threadIdx.x >> 5;
  const int lane  = threadIdx.x & 31;
  const int c0    = wave * 16;
  const int mrow  = lane & 15;
  const int khalf = lane >> 4;

  v8f acc = gemm_k128(&sA[mrow * KPAD], &sWT[(c0 + mrow) * KPAD], khalf);

  // block-uniform camera split (SALU): tile is entirely inside one camera
  const int s  = r0 / MTOT;
  const int m0 = r0 % MTOT;
  const int mb = khalf * 8;
  const int cc = c0 + mrow;
  const int h  = cc >> 5, d = cc & 31;
  const float bias = bv[cc];
  float* outp = val_ws + (((size_t)(s * NHEADS + h) * MTOT) + m0 + mb) * DH + d;
#pragma unroll
  for (int v = 0; v < 8; ++v)
    outp[v * DH] = acc[v] + bias;               // immediate-offset stores
}

// ============================================================================
// Kernel B (chunked): [off_raw | attn_raw] = (query+query_pos) @ [Woff|Wattn]
// + bias. One launch per <=128-column chunk; camera-invariant -> computed once.
// ============================================================================
__global__ void __launch_bounds__(256)
sca_qproj(const float* __restrict__ query, const float* __restrict__ query_pos,
          const float* __restrict__ Woff,  const float* __restrict__ boff,
          const float* __restrict__ Wattn, const float* __restrict__ battn,
          float* __restrict__ off_raw,     // (N, 160)
          float* __restrict__ attn_raw,    // (N, 80)
          int colBase, int nTiles) {
  __shared__ __bf16 sWT[CDIM * KPAD];
  __shared__ __bf16 sA[16 * KPAD];
  const int r0 = blockIdx.x * 16;

  for (int i = threadIdx.x; i < CDIM * (CDIM / 2); i += 256) {
    const int c  = i & (CDIM - 1);
    const int kp = i >> 7;
    const int cc = colBase + c;
    if (cc < 240) {
      const float f0 = (cc < 160) ? Woff[(2 * kp) * 160 + cc]
                                  : Wattn[(2 * kp) * 80 + (cc - 160)];
      const float f1 = (cc < 160) ? Woff[(2 * kp + 1) * 160 + cc]
                                  : Wattn[(2 * kp + 1) * 80 + (cc - 160)];
      const unsigned p = pack2bf(f0, f1);
      __builtin_memcpy(&sWT[c * KPAD + 2 * kp], &p, 4);
    }
  }
  for (int i = threadIdx.x; i < 16 * (CDIM / 2); i += 256) {
    const int r  = i >> 6;
    const int kp = i & 63;
    const size_t base = (size_t)(r0 + r) * CDIM + 2 * kp;
    const float2 f = *(const float2*)&query[base];
    const float2 g = *(const float2*)&query_pos[base];
    const unsigned p = pack2bf(f.x + g.x, f.y + g.y);
    __builtin_memcpy(&sA[r * KPAD + 2 * kp], &p, 4);
  }
  __syncthreads();

  const int wave  = threadIdx.x >> 5;
  const int lane  = threadIdx.x & 31;
  const int mrow  = lane & 15;
  const int khalf = lane >> 4;

  if (wave >= nTiles) return;

  v8f acc = gemm_k128(&sA[mrow * KPAD], &sWT[(wave * 16 + mrow) * KPAD], khalf);

  // select destination stream once per lane (not per element)
  const int mb = khalf * 8;
  const int cc = colBase + wave * 16 + mrow;
  const bool isOff  = (cc < 160);
  const float bias  = isOff ? boff[cc] : battn[cc - 160];
  const int stride  = isOff ? 160 : 80;
  float* dst = (isOff ? off_raw + cc : attn_raw + (cc - 160))
             + (size_t)(r0 + mb) * stride;
#pragma unroll
  for (int v = 0; v < 8; ++v)
    dst[v * stride] = acc[v] + bias;
}

// ============================================================================
// Kernel C: per-(query,head) wave32: softmax(20) + masked 6-cam x 20-pt bilinear
// gather-blend, masked mean -> slots (N, C). lane == channel (DH == 32).
// gid pinned to SGPR so control loads go down the scalar path.
// ============================================================================
__global__ void __launch_bounds__(256)
sca_sample(const float* __restrict__ val_ws,    // (S, HEADS, M, DH)
           const float* __restrict__ off_raw,   // (N, 160)
           const float* __restrict__ attn_raw,  // (N, 80)
           const float* __restrict__ refpts,    // (S, 1, N, D, 2)
           const unsigned char* __restrict__ bev_mask, // (S, 1, N, D) bool
           float* __restrict__ slots) {         // (N, C)
  const int gid  = __builtin_amdgcn_readfirstlane(blockIdx.x * 8 + (threadIdx.x >> 5));
  const int lane = threadIdx.x & 31;
  const int n = gid >> 2;
  const int h = gid & 3;
  if (n >= NQ) return;

  // softmax over the 20 attention logits (wave-uniform scalars)
  const float* ar = attn_raw + (size_t)n * 80 + h * NPOINTS;
  float aw[NPOINTS];
  float mx = -3.0e38f;
#pragma unroll
  for (int p = 0; p < NPOINTS; ++p) { aw[p] = ar[p]; mx = fmaxf(mx, aw[p]); }
  float se = 0.f;
#pragma unroll
  for (int p = 0; p < NPOINTS; ++p) { aw[p] = __expf(aw[p] - mx); se += aw[p]; }
  const float inv = 1.0f / se;
#pragma unroll
  for (int p = 0; p < NPOINTS; ++p) aw[p] *= inv;

  // camera-invariant normalized offsets, hoisted out of the camera loop
  const float* offr = off_raw + (size_t)n * 160 + h * 40;
  float oxv[NPOINTS], oyv[NPOINTS];
#pragma unroll
  for (int p = 0; p < NPOINTS; ++p) {
    const int pg = p >> 2, dz = p & 3;
    oxv[p] = offr[pg * 8 + dz * 2 + 0] * (1.0f / (float)WS_);
    oyv[p] = offr[pg * 8 + dz * 2 + 1] * (1.0f / (float)HS_);
  }

  float acc = 0.f;
  int count = 0;

  for (int s = 0; s < SCAMS; ++s) {
    unsigned m4;
    __builtin_memcpy(&m4, bev_mask + ((size_t)s * NQ + n) * DZ, 4);
    if (m4 == 0u) continue;                     // scalar branch
    ++count;
    const float* img  = val_ws + ((size_t)(s * NHEADS + h) * MTOT) * DH;
    const float* refc = refpts + ((size_t)s * NQ + n) * (DZ * 2);
    __builtin_prefetch(img, 0, 1);
#pragma unroll
    for (int p = 0; p < NPOINTS; ++p) {
      const int dz = p & 3;
      const float x = (refc[dz * 2 + 0] + oxv[p]) * (float)WS_ - 0.5f;
      const float y = (refc[dz * 2 + 1] + oyv[p]) * (float)HS_ - 0.5f;
      const float x0f = floorf(x), y0f = floorf(y);
      const int x0 = (int)x0f, y0 = (int)y0f;
      const float wx1 = x - x0f, wy1 = y - y0f;
      const float wx0 = 1.0f - wx1, wy0 = 1.0f - wy1;
      const bool xi0 = (x0 >= 0) && (x0 < WS_);
      const bool xi1 = (x0 + 1 >= 0) && (x0 + 1 < WS_);
      const bool yi0 = (y0 >= 0) && (y0 < HS_);
      const bool yi1 = (y0 + 1 >= 0) && (y0 + 1 < HS_);
      float v00 = 0.f, v01 = 0.f, v10 = 0.f, v11 = 0.f;
      if (yi0) {
        const float* row = img + (size_t)y0 * WS_ * DH;
        if (xi0) v00 = row[x0 * DH + lane];     // 128B coalesced wave load
        if (xi1) v01 = row[(x0 + 1) * DH + lane];
      }
      if (yi1) {
        const float* row = img + (size_t)(y0 + 1) * WS_ * DH;
        if (xi0) v10 = row[x0 * DH + lane];
        if (xi1) v11 = row[(x0 + 1) * DH + lane];
      }
      acc += aw[p] * ((v00 * wx0 + v01 * wx1) * wy0 + (v10 * wx0 + v11 * wx1) * wy1);
    }
  }
  const float scale = 1.0f / fmaxf((float)count, 1.0f);
  slots[(size_t)n * CDIM + h * DH + lane] = acc * scale;
}

// ============================================================================
// Kernel D: out = slots @ Wout + bout + query (residual)
// ============================================================================
__global__ void __launch_bounds__(256)
sca_out_proj(const float* __restrict__ slots, const float* __restrict__ Wout,
             const float* __restrict__ bout,  const float* __restrict__ query,
             float* __restrict__ out) {
  __shared__ __bf16 sWT[CDIM * KPAD];
  __shared__ __bf16 sA[16 * KPAD];
  const int r0 = blockIdx.x * 16;

  for (int i = threadIdx.x; i < CDIM * (CDIM / 2); i += 256) {
    const int c  = i & (CDIM - 1);
    const int kp = i >> 7;
    const unsigned p = pack2bf(Wout[(2 * kp) * CDIM + c], Wout[(2 * kp + 1) * CDIM + c]);
    __builtin_memcpy(&sWT[c * KPAD + 2 * kp], &p, 4);
  }
  for (int i = threadIdx.x; i < 16 * (CDIM / 2); i += 256) {
    const int r  = i >> 6;
    const int kp = i & 63;
    const float2 f = *(const float2*)&slots[(size_t)(r0 + r) * CDIM + 2 * kp];
    const unsigned p = pack2bf(f.x, f.y);
    __builtin_memcpy(&sA[r * KPAD + 2 * kp], &p, 4);
  }
  __syncthreads();

  const int wave  = threadIdx.x >> 5;
  const int lane  = threadIdx.x & 31;
  const int c0    = wave * 16;
  const int mrow  = lane & 15;
  const int khalf = lane >> 4;

  v8f acc = gemm_k128(&sA[mrow * KPAD], &sWT[(c0 + mrow) * KPAD], khalf);

  const int mb = khalf * 8;
  const int cc = c0 + mrow;
  const float bias = bout[cc];
  const float* resp = query + (size_t)(r0 + mb) * CDIM + cc;
  float* outp = out + (size_t)(r0 + mb) * CDIM + cc;
#pragma unroll
  for (int v = 0; v < 8; ++v)
    outp[v * CDIM] = acc[v] + bias + resp[v * CDIM];
}

// ============================================================================
extern "C" void kernel_launch(void* const* d_in, const int* in_sizes, int n_in,
                              void* d_out, int out_size, void* d_ws, size_t ws_size,
                              hipStream_t stream) {
  const float* query     = (const float*)d_in[0];   // (1, N, C)
  // d_in[1] = key (unused by the reference computation)
  const float* value     = (const float*)d_in[2];   // (S, M, 1, C)
  const float* query_pos = (const float*)d_in[3];   // (1, N, C)
  const float* refpts    = (const float*)d_in[4];   // (S, 1, N, D, 2)
  const unsigned char* bev_mask = (const unsigned char*)d_in[5]; // (S,1,N,D) bool
  // d_in[6] spatial_shapes, d_in[7] level_start_index: compile-time constants here
  const float* Wv    = (const float*)d_in[8];
  const float* bv    = (const float*)d_in[9];
  const float* Woff  = (const float*)d_in[10];
  const float* boff  = (const float*)d_in[11];
  const float* Wattn = (const float*)d_in[12];
  const float* battn = (const float*)d_in[13];
  const float* Wout  = (const float*)d_in[14];
  const float* bout  = (const float*)d_in[15];
  float* out = (float*)d_out;

  char* ws = (char*)d_ws;
  size_t off = 0;
  auto take = [&](size_t bytes) {
    char* p = ws + off;
    off = (off + bytes + 255) & ~(size_t)255;
    return p;
  };
  float* val_ws   = (float*)take((size_t)SCAMS * NHEADS * MTOT * DH * sizeof(float));
  float* off_raw  = (float*)take((size_t)NQ * 160 * sizeof(float));
  float* attn_raw = (float*)take((size_t)NQ * 80 * sizeof(float));
  float* slots    = (float*)take((size_t)NQ * CDIM * sizeof(float));
  (void)ws_size; (void)in_sizes; (void)n_in; (void)out_size;

  // A: value projection -> per-(camera,head) images
  sca_val_proj<<<(SCAMS * MTOT) / 16, 256, 0, stream>>>(value, Wv, bv, val_ws);
  // B: camera-invariant offset/attention projection (two column chunks)
  sca_qproj<<<NQ / 16, 256, 0, stream>>>(query, query_pos, Woff, boff,
                                         Wattn, battn, off_raw, attn_raw, 0, 8);
  sca_qproj<<<NQ / 16, 256, 0, stream>>>(query, query_pos, Woff, boff,
                                         Wattn, battn, off_raw, attn_raw, 128, 7);
  // C: masked multi-camera deformable sampling -> slots
  sca_sample<<<(NQ * NHEADS) / 8, 256, 0, stream>>>(val_ws, off_raw, attn_raw,
                                                    refpts, bev_mask, slots);
  // D: output projection + residual
  sca_out_proj<<<NQ / 16, 256, 0, stream>>>(slots, Wout, bout, query, out);
}
